// KernelCouplingA_CausalPool_15324443312609
// MI455X (gfx1250) — compile-verified, matrix operand order checked by reading, not run on hardware
//
#include <hip/hip_runtime.h>
#include <hip/hip_bf16.h>
#include <math.h>

// ---------------------------------------------------------------------------
// Problem constants (from reference)
// ---------------------------------------------------------------------------
#define B_   2
#define L_   2048
#define D_   1024
#define NH_  16
#define HD_  64       // D_/NH_
#define HG_  256      // H_GATE
#define LOG2_FAST 0.6942419136306174f   // log2(phi)
#define LOG2_SLOW 10.660026f            // log2(1618.0)

typedef unsigned short ushort_t;
typedef __attribute__((ext_vector_type(16))) __bf16          v16bf;
typedef __attribute__((ext_vector_type(8)))  float           v8f;
typedef __attribute__((ext_vector_type(8)))  unsigned short  ushort8;
typedef __attribute__((ext_vector_type(4)))  unsigned int    v4u;
typedef __attribute__((ext_vector_type(8)))  int             v8i;
typedef __attribute__((ext_vector_type(4)))  int             v4i;

union FragBF16 {
  v16bf   v;
  ushort8 h[2];
};

#if defined(__has_builtin)
#  if __has_builtin(__builtin_amdgcn_tensor_load_to_lds)
#    define HAVE_TDM 1
#  else
#    define HAVE_TDM 0
#  endif
#  if __has_builtin(__builtin_amdgcn_s_wait_tensorcnt)
#    define WAIT_TENSORCNT0() __builtin_amdgcn_s_wait_tensorcnt(0)
#  else
#    define WAIT_TENSORCNT0() asm volatile("s_wait_tensorcnt 0x0" ::: "memory")
#  endif
#else
#  define HAVE_TDM 0
#  define WAIT_TENSORCNT0() asm volatile("s_wait_tensorcnt 0x0" ::: "memory")
#endif

__device__ __forceinline__ unsigned short f2bf(float f) {
  unsigned int u = __float_as_uint(f);
  unsigned int r = u + 0x7FFFu + ((u >> 16) & 1u);   // round-to-nearest-even
  return (unsigned short)(r >> 16);
}

#if HAVE_TDM
// Issue a TDM 2D tile load (bf16 elements): tile_h rows x tile_w elems,
// global row stride = row_stride_elems, into LDS at byte offset lds_off.
__device__ __forceinline__ void tdm_load_2d_bf16(unsigned lds_off, const void* gptr,
                                                 unsigned tile_w, unsigned tile_h,
                                                 unsigned row_stride_elems) {
  unsigned long long ga = (unsigned long long)gptr;
  v4u g0;
  g0.x = 1u;                                               // count=1, user D#
  g0.y = lds_off;                                          // lds_addr
  g0.z = (unsigned)ga;                                     // global_addr[31:0]
  g0.w = (unsigned)((ga >> 32) & 0x1FFFFFFu) | (2u << 30); // addr[56:32] | type=2
  v8i g1;
  g1[0] = 0x00010000;                 // workgroup_mask=0, data_size=1 (2 bytes)
  g1[1] = (int)(tile_w << 16);        // tensor_dim0[15:0] @ bits 63:48
  g1[2] = (int)(tile_h << 16);        // tensor_dim1[15:0] @ bits 95:80
  g1[3] = (int)(tile_w << 16);        // tile_dim0 @ bits 127:112
  g1[4] = (int)tile_h;                // tile_dim1 @ bits 143:128
  g1[5] = (int)row_stride_elems;      // tensor_dim0_stride[31:0]
  g1[6] = 0;
  g1[7] = 0;
  v4i z4 = {0, 0, 0, 0};
#if __clang_major__ >= 23
  v8i z8 = {0, 0, 0, 0, 0, 0, 0, 0};
  __builtin_amdgcn_tensor_load_to_lds(g0, g1, z4, z4, z8, 0);
#else
  __builtin_amdgcn_tensor_load_to_lds(g0, g1, z4, z4, 0);
#endif
}
#endif

// ---------------------------------------------------------------------------
// 1) f32 -> bf16 conversion
// ---------------------------------------------------------------------------
__global__ void conv_bf16_kernel(const float* __restrict__ in,
                                 ushort_t* __restrict__ out, long n) {
  long i = (long)blockIdx.x * blockDim.x + threadIdx.x;
  if (i < n) out[i] = f2bf(in[i]);
}

// ---------------------------------------------------------------------------
// 2) bf16 WMMA GEMM:  C[M,N] = (A[M,K] @ B[N,K]^T) * smul + bias[N]
//    smul = scale * tanh(*gamma) when gamma != nullptr, else scale.
//    Block = 256 threads = 8 waves; wave computes a 32x64 tile
//    (block tile = 256 rows x 64 cols). The 64x32 B-tile per k-step is
//    staged into LDS once per block. With TDM available the tiles are
//    DOUBLE-BUFFERED: wave 0 issues the DMA for tile k+1 right after the
//    top barrier, then the block computes tile k, so the tensor DMA
//    overlaps the 8 WMMAs + A-loads of the compute phase.
//    Grid = (N/64, M/256, batches).
// ---------------------------------------------------------------------------
__global__ __launch_bounds__(256)
void gemm_bf16_nt(const ushort_t* __restrict__ A, const ushort_t* __restrict__ B,
                  const float* __restrict__ bias, float* __restrict__ C,
                  int M, int N, int K,
                  long strideA, long strideB, long strideC,
                  float scale, const float* __restrict__ gamma) {
  (void)M;
  __shared__ ushort_t Btile[2][64 * 32];   // 2 x 4 KB: 64 rows x 32 k, row-major
  const int bat  = blockIdx.z;
  A += (long)bat * strideA;
  B += (long)bat * strideB;
  C += (long)bat * strideC;

  const int tid  = threadIdx.x;
  const int wave = tid >> 5;
  const int lane = tid & 31;
  const int half = lane >> 4;
  const int l15  = lane & 15;
  const int m0   = blockIdx.y * 256 + wave * 32;
  const int n0   = blockIdx.x * 64;
  const int kA0  = half ? 8 : 0;       // per-lane K chunk offsets (bf16 A/B layout)
  const int kA1  = half ? 24 : 16;

  v8f acc[2][4];
#pragma unroll
  for (int g = 0; g < 2; ++g)
#pragma unroll
    for (int t = 0; t < 4; ++t) acc[g][t] = v8f{0};

  const ushort_t* arow0 = A + (long)(m0 + l15) * K;
  const ushort_t* arow1 = A + (long)(m0 + 16 + l15) * K;
  const ushort_t* brow0 = B + (long)n0 * K;

  const int nk = K / 32;

#if HAVE_TDM
  // Prologue: kick off tile 0 into buffer 0.
  if (wave == 0) {
    tdm_load_2d_bf16((unsigned)(size_t)(void*)&Btile[0][0], brow0, 32u, 64u,
                     (unsigned)K);
  }
#endif

  for (int ks = 0; ks < nk; ++ks) {
    const int k = ks * 32;
#if HAVE_TDM
    if (wave == 0) WAIT_TENSORCNT0();    // tile ks has landed in LDS
    __syncthreads();                     // ...and buffer (ks+1)&1 fully read
    if (wave == 0 && ks + 1 < nk) {      // overlap DMA of tile ks+1 w/ compute
      tdm_load_2d_bf16((unsigned)(size_t)(void*)&Btile[(ks + 1) & 1][0],
                       brow0 + k + 32, 32u, 64u, (unsigned)K);
    }
    const ushort_t* bt = &Btile[ks & 1][0];
#else
    __syncthreads();                     // previous iteration's reads complete
    {
      int r = tid >> 2, c = tid & 3;     // 64 rows x 4 chunks of 16B
      *(ushort8*)(&Btile[0][r * 32 + c * 8]) =
          *(const ushort8*)(brow0 + (long)r * K + k + c * 8);
    }
    __syncthreads();
    const ushort_t* bt = &Btile[0][0];
#endif

    __builtin_prefetch(arow0 + k + 128, 0, 3);
    __builtin_prefetch(arow1 + k + 128, 0, 3);
    FragBF16 a0, a1;
    a0.h[0] = *(const ushort8*)(arow0 + k + kA0);
    a0.h[1] = *(const ushort8*)(arow0 + k + kA1);
    a1.h[0] = *(const ushort8*)(arow1 + k + kA0);
    a1.h[1] = *(const ushort8*)(arow1 + k + kA1);
#pragma unroll
    for (int t = 0; t < 4; ++t) {
      const ushort_t* brow = bt + (t * 16 + l15) * 32;
      FragBF16 bf;
      bf.h[0] = *(const ushort8*)(brow + kA0);
      bf.h[1] = *(const ushort8*)(brow + kA1);
      acc[0][t] = __builtin_amdgcn_wmma_f32_16x16x32_bf16(
          false, a0.v, false, bf.v, (short)0, acc[0][t], false, false);
      acc[1][t] = __builtin_amdgcn_wmma_f32_16x16x32_bf16(
          false, a1.v, false, bf.v, (short)0, acc[1][t], false, false);
    }
#if HAVE_TDM
    __syncthreads();   // reads of buffer ks&1 done before it is re-filled
#endif
  }

  float smul = scale * (gamma ? tanhf(gamma[0]) : 1.0f);
#pragma unroll
  for (int g = 0; g < 2; ++g) {
#pragma unroll
    for (int t = 0; t < 4; ++t) {
      int nn = n0 + t * 16 + l15;
      float bs = bias ? bias[nn] : 0.0f;
#pragma unroll
      for (int r = 0; r < 8; ++r) {
        int mm = m0 + g * 16 + 8 * half + r;
        C[(long)mm * N + nn] = acc[g][t][r] * smul + bs;
      }
    }
  }
}

// ---------------------------------------------------------------------------
// 3) Gate: gate[m] = sigmoid( gelu(h[m,:]) . Wg2 + bg2 )
// ---------------------------------------------------------------------------
__global__ void gate_kernel(const float* __restrict__ h,
                            const float* __restrict__ Wg2,
                            const float* __restrict__ bg2,
                            float* __restrict__ gate, int rows) {
  int i = blockIdx.x * blockDim.x + threadIdx.x;
  if (i >= rows) return;
  const float* hr = h + (long)i * HG_;
  float acc = bg2[0];
  for (int j = 0; j < HG_; ++j) {
    float x = hr[j];
    float g = 0.5f * x * (1.0f + erff(x * 0.7071067811865475f));
    acc += g * Wg2[j];
  }
  gate[i] = 1.0f / (1.0f + __expf(-acc));
}

// ---------------------------------------------------------------------------
// 4) Gated dual-base RoPE + bf16 re-layout.
//    Writes: Qrb, Krb (rotated, [b,l,d]); Krawb, gkb (plain bf16 [b,l,d]);
//            Vtb transposed to [b, d, l] so PV B-fragments are contiguous.
// ---------------------------------------------------------------------------
__global__ void rope_kernel(const float* __restrict__ Qf, const float* __restrict__ Kf,
                            const float* __restrict__ Vf, const float* __restrict__ gkf,
                            const float* __restrict__ gate,
                            ushort_t* __restrict__ Qrb, ushort_t* __restrict__ Krb,
                            ushort_t* __restrict__ Krawb, ushort_t* __restrict__ gkb,
                            ushort_t* __restrict__ Vtb) {
  const int row = blockIdx.x;            // 0 .. B*L-1
  const int b = row / L_, l = row % L_;
  const float g  = gate[row];
  const float t  = (float)l;
  for (int j = threadIdx.x; j < D_; j += blockDim.x) {
    int   i    = j & (D_ / 2 - 1);
    float frac = (float)i * (2.0f / (float)D_);       // i / 512
    float af   = t * exp2f(-frac * LOG2_FAST);
    float as   = t * exp2f(-frac * LOG2_SLOW);
    float sf, cf, ss, cs;
    sincosf(af, &sf, &cf);
    sincosf(as, &ss, &cs);
    float cosm = g * cf + (1.0f - g) * cs;
    float sinm = g * sf + (1.0f - g) * ss;

    long idx  = (long)row * D_ + j;
    long pidx = (long)row * D_ + ((j < D_ / 2) ? (j + D_ / 2) : (j - D_ / 2));
    float sign = (j < D_ / 2) ? -1.0f : 1.0f;

    float q  = Qf[idx], k = Kf[idx];
    float qh = sign * Qf[pidx];
    float kh = sign * Kf[pidx];

    Qrb[idx]   = f2bf(q * cosm + qh * sinm);
    Krb[idx]   = f2bf(k * cosm + kh * sinm);
    Krawb[idx] = f2bf(k);
    gkb[idx]   = f2bf(gkf[idx]);
    Vtb[((long)b * D_ + j) * L_ + l] = f2bf(Vf[idx]);
  }
}

// ---------------------------------------------------------------------------
// 5) Causal attention with gb bias, streaming two-pass softmax.
//    Grid = (L/64, NH, B), 128 threads (4 waves); wave owns 16 query rows.
// ---------------------------------------------------------------------------
__global__ __launch_bounds__(128)
void attn_kernel(const ushort_t* __restrict__ Qrb, const ushort_t* __restrict__ Krb,
                 const ushort_t* __restrict__ Vtb, const float* __restrict__ gb,
                 float* __restrict__ out, float* __restrict__ attnw) {
  __shared__ ushort_t Pbuf[4][16 * 64];

  const int wave = threadIdx.x >> 5;
  const int lane = threadIdx.x & 31;
  const int half = lane >> 4;
  const int l15  = lane & 15;
  const int h    = blockIdx.y, b = blockIdx.z;
  const int qtile = blockIdx.x;
  const int qbase = qtile * 64 + wave * 16;
  const int numkt = qtile + 1;
  const int kA0 = half ? 8 : 0;
  const int kA1 = half ? 24 : 16;
  const float scale = 0.125f;            // 1/sqrt(HD)

  // Q fragments (16 rows x 64 head-dim) -- resident for whole kernel.
  const ushort_t* Qrow = Qrb + ((long)(b * L_ + qbase + l15)) * D_ + h * HD_;
  FragBF16 aq[2];
  aq[0].h[0] = *(const ushort8*)(Qrow + kA0);
  aq[0].h[1] = *(const ushort8*)(Qrow + kA1);
  aq[1].h[0] = *(const ushort8*)(Qrow + 32 + kA0);
  aq[1].h[1] = *(const ushort8*)(Qrow + 32 + kA1);

  const float* gbb   = gb    + ((long)b * L_) * L_;
  float*       attnb = attnw + ((long)b * L_) * L_;

  float mrun[8], srun[8];
#pragma unroll
  for (int r = 0; r < 8; ++r) { mrun[r] = -3.0e38f; srun[r] = 0.0f; }

  // ---------------- Pass 1: row max + sum of exp ----------------
  for (int kt = 0; kt < numkt; ++kt) {
    v8f s[4] = {v8f{0}, v8f{0}, v8f{0}, v8f{0}};
#pragma unroll
    for (int t = 0; t < 4; ++t) {
      const ushort_t* Krow = Krb + ((long)(b * L_ + kt * 64 + t * 16 + l15)) * D_ + h * HD_;
      FragBF16 b0, b1;
      b0.h[0] = *(const ushort8*)(Krow + kA0);
      b0.h[1] = *(const ushort8*)(Krow + kA1);
      b1.h[0] = *(const ushort8*)(Krow + 32 + kA0);
      b1.h[1] = *(const ushort8*)(Krow + 32 + kA1);
      s[t] = __builtin_amdgcn_wmma_f32_16x16x32_bf16(false, aq[0].v, false, b0.v,
                                                     (short)0, s[t], false, false);
      s[t] = __builtin_amdgcn_wmma_f32_16x16x32_bf16(false, aq[1].v, false, b1.v,
                                                     (short)0, s[t], false, false);
    }
    float tmax[8];
#pragma unroll
    for (int r = 0; r < 8; ++r) tmax[r] = -3.0e38f;
#pragma unroll
    for (int t = 0; t < 4; ++t) {
      int kk = kt * 64 + t * 16 + l15;
#pragma unroll
      for (int r = 0; r < 8; ++r) {
        int qq = qbase + 8 * half + r;
        float v = s[t][r] * scale + gbb[(long)qq * L_ + kk];
        if (kk > qq) v = -3.0e38f;
        s[t][r] = v;
        tmax[r] = fmaxf(tmax[r], v);
      }
    }
#pragma unroll
    for (int r = 0; r < 8; ++r) {
      float v = tmax[r];
      for (int m = 8; m >= 1; m >>= 1) v = fmaxf(v, __shfl_xor(v, m, 32));
      float mnew = fmaxf(mrun[r], v);
      float ssum = 0.0f;
#pragma unroll
      for (int t = 0; t < 4; ++t) ssum += __expf(s[t][r] - mnew);
      for (int m = 8; m >= 1; m >>= 1) ssum += __shfl_xor(ssum, m, 32);
      srun[r] = srun[r] * __expf(mrun[r] - mnew) + ssum;
      mrun[r] = mnew;
    }
  }
  float linv[8];
#pragma unroll
  for (int r = 0; r < 8; ++r) linv[r] = 1.0f / srun[r];

  // ---------------- Pass 2: P, attn mean, O = P @ V ----------------
  v8f o[4] = {v8f{0}, v8f{0}, v8f{0}, v8f{0}};
  for (int kt = 0; kt < numkt; ++kt) {
    v8f s[4] = {v8f{0}, v8f{0}, v8f{0}, v8f{0}};
#pragma unroll
    for (int t = 0; t < 4; ++t) {
      const ushort_t* Krow = Krb + ((long)(b * L_ + kt * 64 + t * 16 + l15)) * D_ + h * HD_;
      FragBF16 b0, b1;
      b0.h[0] = *(const ushort8*)(Krow + kA0);
      b0.h[1] = *(const ushort8*)(Krow + kA1);
      b1.h[0] = *(const ushort8*)(Krow + 32 + kA0);
      b1.h[1] = *(const ushort8*)(Krow + 32 + kA1);
      s[t] = __builtin_amdgcn_wmma_f32_16x16x32_bf16(false, aq[0].v, false, b0.v,
                                                     (short)0, s[t], false, false);
      s[t] = __builtin_amdgcn_wmma_f32_16x16x32_bf16(false, aq[1].v, false, b1.v,
                                                     (short)0, s[t], false, false);
    }
    __syncthreads();   // protect Pbuf of previous iteration
#pragma unroll
    for (int t = 0; t < 4; ++t) {
      int kk = kt * 64 + t * 16 + l15;
#pragma unroll
      for (int r = 0; r < 8; ++r) {
        int qq = qbase + 8 * half + r;
        float v = s[t][r] * scale + gbb[(long)qq * L_ + kk];
        float p = 0.0f;
        if (kk <= qq) {
          p = __expf(v - mrun[r]) * linv[r];
          atomicAdd(&attnb[(long)qq * L_ + kk], p * (1.0f / (float)NH_));
        }
        Pbuf[wave][(r + 8 * half) * 64 + t * 16 + l15] = f2bf(p);
      }
    }
    __syncthreads();
    const ushort_t* pw = &Pbuf[wave][l15 * 64];
    FragBF16 pa[2];
    pa[0].h[0] = *(const ushort8*)(pw + kA0);
    pa[0].h[1] = *(const ushort8*)(pw + kA1);
    pa[1].h[0] = *(const ushort8*)(pw + 32 + kA0);
    pa[1].h[1] = *(const ushort8*)(pw + 32 + kA1);
#pragma unroll
    for (int t = 0; t < 4; ++t) {
      const ushort_t* Vrow = Vtb + ((long)b * D_ + h * HD_ + t * 16 + l15) * L_ + kt * 64;
      FragBF16 v0, v1;
      v0.h[0] = *(const ushort8*)(Vrow + kA0);
      v0.h[1] = *(const ushort8*)(Vrow + kA1);
      v1.h[0] = *(const ushort8*)(Vrow + 32 + kA0);
      v1.h[1] = *(const ushort8*)(Vrow + 32 + kA1);
      o[t] = __builtin_amdgcn_wmma_f32_16x16x32_bf16(false, pa[0].v, false, v0.v,
                                                     (short)0, o[t], false, false);
      o[t] = __builtin_amdgcn_wmma_f32_16x16x32_bf16(false, pa[1].v, false, v1.v,
                                                     (short)0, o[t], false, false);
    }
  }
#pragma unroll
  for (int t = 0; t < 4; ++t) {
#pragma unroll
    for (int r = 0; r < 8; ++r) {
      int qq = qbase + 8 * half + r;
      out[((long)(b * L_ + qq)) * D_ + h * HD_ + t * 16 + l15] = o[t][r];
    }
  }
}

// ---------------------------------------------------------------------------
// Launcher
// ---------------------------------------------------------------------------
extern "C" void kernel_launch(void* const* d_in, const int* in_sizes, int n_in,
                              void* d_out, int out_size, void* d_ws, size_t ws_size,
                              hipStream_t stream) {
  (void)in_sizes; (void)n_in; (void)out_size; (void)ws_size;
  const float* z     = (const float*)d_in[0];
  const float* pool  = (const float*)d_in[1];
  const float* Wq    = (const float*)d_in[2];
  const float* bq    = (const float*)d_in[3];
  const float* Wk    = (const float*)d_in[4];
  const float* bk    = (const float*)d_in[5];
  const float* Wv    = (const float*)d_in[6];
  const float* bv    = (const float*)d_in[7];
  const float* Wcoh  = (const float*)d_in[8];
  const float* gamma = (const float*)d_in[9];
  const float* Wg1   = (const float*)d_in[10];
  const float* bg1   = (const float*)d_in[11];
  const float* Wg2   = (const float*)d_in[12];
  const float* bg2   = (const float*)d_in[13];

  const long BLD = (long)B_ * L_ * D_;     // 4,194,304
  const long BLL = (long)B_ * L_ * L_;     // 8,388,608
  const long DD  = (long)D_ * D_;

  char* w = (char*)d_ws;
  auto carve = [&](size_t bytes) -> char* {
    char* p = w;
    w += (bytes + 255) & ~(size_t)255;
    return p;
  };
  ushort_t* zb    = (ushort_t*)carve(BLD * 2);
  ushort_t* pb    = (ushort_t*)carve(BLD * 2);
  ushort_t* wqb   = (ushort_t*)carve(DD * 2);
  ushort_t* wkb   = (ushort_t*)carve(DD * 2);
  ushort_t* wvb   = (ushort_t*)carve(DD * 2);
  ushort_t* wcb   = (ushort_t*)carve(DD * 2);
  ushort_t* wg1b  = (ushort_t*)carve((long)HG_ * D_ * 2);
  float*    Qf    = (float*)carve(BLD * 4);
  float*    Kf    = (float*)carve(BLD * 4);
  float*    Vf    = (float*)carve(BLD * 4);
  float*    gkf   = (float*)carve(BLD * 4);
  float*    hbuf  = (float*)carve((long)B_ * L_ * HG_ * 4);
  float*    gate  = (float*)carve((long)B_ * L_ * 4);
  ushort_t* Qrb   = (ushort_t*)carve(BLD * 2);
  ushort_t* Krb   = (ushort_t*)carve(BLD * 2);
  ushort_t* Krawb = (ushort_t*)carve(BLD * 2);
  ushort_t* gkb   = (ushort_t*)carve(BLD * 2);
  ushort_t* Vtb   = (ushort_t*)carve(BLD * 2);
  float*    gbb   = (float*)carve(BLL * 4);

  float* out_ptr  = (float*)d_out;
  float* attn_ptr = out_ptr + BLD;

  // --- 1) convert inputs/weights to bf16 ---
  auto cgrid = [](long n) { return (unsigned)((n + 255) / 256); };
  conv_bf16_kernel<<<cgrid(BLD), 256, 0, stream>>>(z, zb, BLD);
  conv_bf16_kernel<<<cgrid(BLD), 256, 0, stream>>>(pool, pb, BLD);
  conv_bf16_kernel<<<cgrid(DD), 256, 0, stream>>>(Wq, wqb, DD);
  conv_bf16_kernel<<<cgrid(DD), 256, 0, stream>>>(Wk, wkb, DD);
  conv_bf16_kernel<<<cgrid(DD), 256, 0, stream>>>(Wv, wvb, DD);
  conv_bf16_kernel<<<cgrid(DD), 256, 0, stream>>>(Wcoh, wcb, DD);
  conv_bf16_kernel<<<cgrid((long)HG_ * D_), 256, 0, stream>>>(Wg1, wg1b, (long)HG_ * D_);

  // --- 2) projection GEMMs:  [4096,1024] = [4096,1024] @ [1024,1024]^T ---
  const int Mrows = B_ * L_;   // 4096
  gemm_bf16_nt<<<dim3(D_ / 64, Mrows / 256, 1), 256, 0, stream>>>(
      zb, wqb, bq, Qf, Mrows, D_, D_, 0, 0, 0, 1.0f, nullptr);
  gemm_bf16_nt<<<dim3(D_ / 64, Mrows / 256, 1), 256, 0, stream>>>(
      zb, wkb, bk, Kf, Mrows, D_, D_, 0, 0, 0, 1.0f, nullptr);
  gemm_bf16_nt<<<dim3(D_ / 64, Mrows / 256, 1), 256, 0, stream>>>(
      zb, wvb, bv, Vf, Mrows, D_, D_, 0, 0, 0, 1.0f, nullptr);
  gemm_bf16_nt<<<dim3(D_ / 64, Mrows / 256, 1), 256, 0, stream>>>(
      pb, wcb, nullptr, gkf, Mrows, D_, D_, 0, 0, 0, 1.0f, nullptr);
  gemm_bf16_nt<<<dim3(HG_ / 64, Mrows / 256, 1), 256, 0, stream>>>(
      pb, wg1b, bg1, hbuf, Mrows, HG_, D_, 0, 0, 0, 1.0f, nullptr);

  // --- 3) gate ---
  gate_kernel<<<(Mrows + 255) / 256, 256, 0, stream>>>(hbuf, Wg2, bg2, gate, Mrows);

  // --- 4) RoPE + bf16 relayouts ---
  rope_kernel<<<Mrows, 256, 0, stream>>>(Qf, Kf, Vf, gkf, gate,
                                         Qrb, Krb, Krawb, gkb, Vtb);

  // --- 5) gb = tanh(gamma)/sqrt(D) * gk @ K_raw^T  (batched over B) ---
  gemm_bf16_nt<<<dim3(L_ / 64, L_ / 256, B_), 256, 0, stream>>>(
      gkb, Krawb, nullptr, gbb, L_, L_, D_,
      (long)L_ * D_, (long)L_ * D_, (long)L_ * L_,
      0.03125f /* D^-0.5 */, gamma);

  // --- 6) zero attn-mean output region (accumulated via atomics) ---
  hipMemsetAsync(attn_ptr, 0, (size_t)BLL * sizeof(float), stream);

  // --- 7) causal attention ---
  attn_kernel<<<dim3(L_ / 64, NH_, B_), 128, 0, stream>>>(
      Qrb, Krb, Vtb, gbb, out_ptr, attn_ptr);
}
